// GNNModel_65584150610196
// MI455X (gfx1250) — matrix operand order repaired and checked by reading, not compiled, not run on hardware
//
#include <hip/hip_runtime.h>
#include <hip/hip_bf16.h>
#include <math.h>

// CDNA5 / gfx1250 — wave32, WMMA 16x16x32 bf16 with f32 accumulate.
// Weight panels staged in LDS (bank-conflict-padded); B fragments software-pipelined
// across k-steps so ds_load latency overlaps WMMA execution.

typedef __attribute__((ext_vector_type(16))) __bf16 v16bf;
typedef __attribute__((ext_vector_type(8)))  float  v8f;

#define BN_EPS_F 1e-5f

__device__ __forceinline__ float gelu_f(float x) {
    // jax.nn.gelu default (approximate=True, tanh form)
    const float c0 = 0.7978845608028654f; // sqrt(2/pi)
    return 0.5f * x * (1.0f + tanhf(c0 * (x + 0.044715f * x * x * x)));
}

__device__ __forceinline__ void atomic_add_agent(float* p, float v) {
    __hip_atomic_fetch_add(p, v, __ATOMIC_RELAXED, __HIP_MEMORY_SCOPE_AGENT);
}

// ----------------------------- small helpers -----------------------------

__global__ void k_fill(float* __restrict__ p, float v, int n) {
    int i = blockIdx.x * blockDim.x + threadIdx.x;
    for (; i < n; i += gridDim.x * blockDim.x) p[i] = v;
}

// WT[n*K + k] = bf16(W[k*NC + n])  (column-major bf16 weights -> contiguous-K B fragments)
__global__ void k_cvt_transpose_bf16(const float* __restrict__ W, __bf16* __restrict__ WT,
                                     int K, int NC) {
    int i = blockIdx.x * blockDim.x + threadIdx.x;
    const int total = K * NC;
    for (; i < total; i += gridDim.x * blockDim.x) {
        const int k = i / NC;
        const int n = i - k * NC;
        WT[(size_t)n * K + k] = (__bf16)W[i];
    }
}

__global__ void k_deg_count(const int* __restrict__ dst, float* __restrict__ deg, int E) {
    int i = blockIdx.x * blockDim.x + threadIdx.x;
    for (; i < E; i += gridDim.x * blockDim.x)
        atomic_add_agent(&deg[dst[i]], 1.0f);
}

__global__ void k_rsqrt_inplace(float* __restrict__ d, int n) {
    int i = blockIdx.x * blockDim.x + threadIdx.x;
    for (; i < n; i += gridDim.x * blockDim.x)
        d[i] = rsqrtf(fmaxf(d[i], 1.0f));
}

// ----------------------------- WMMA GEMM -----------------------------
// C[nrows, NT*16] = act(A[nrows, K] @ B[K, NT*16] + bias)
// BT is bf16 column-major: BT[n*K + k] = B[k, n].
// The weight panel is staged into LDS in 128-deep K-chunks; each staged column is
// padded to 272B (68 words; 68 mod 64 = 4) so the 16 fragment columns of a half-wave
// hit 16 distinct banks while keeping 16B alignment for ds_load_b128.
// One wave per 16-row M-tile; NT f32 accumulator tiles (NT*16 output cols).
// B fragments are double-buffered across k-steps: loads for step k+32 are issued
// before the WMMAs of step k, hiding LDS latency behind matrix-core execution.
template <int NT, int ACT>
__global__ void k_gemm_wmma(const float* __restrict__ A, const __bf16* __restrict__ BT,
                            const float* __restrict__ bias, float* __restrict__ C,
                            int nrows, int K) {
    constexpr int ncols = NT * 16;
    constexpr int KC    = 128;            // K-chunk depth staged in LDS
    constexpr int ROWB  = 2 * KC + 16;    // bytes per staged column (16B pad)
    extern __shared__ char smem[];        // ncols * ROWB bytes (<= 34 KB)

    const int lane = threadIdx.x & 31;
    const int wave = threadIdx.x >> 5;
    const int tile = blockIdx.x * (blockDim.x >> 5) + wave;
    const int m0   = tile << 4;
    const int half = lane >> 4;           // 0..1
    const int sub  = lane & 15;           // A row within tile / B,C column within tile
    const bool active = (m0 < nrows);     // wave-uniform: EXEC stays all-ones for WMMA

    const v8f vzero = {0.f, 0.f, 0.f, 0.f, 0.f, 0.f, 0.f, 0.f};
    v8f acc[NT];
#pragma unroll
    for (int i = 0; i < NT; ++i) acc[i] = vzero;

    // Per-lane base of this lane's fragment column in the staged panel.
    const char* bbase = smem + sub * ROWB + 32 * half;  // (16*half) bf16 = 32B

    for (int kc = 0; kc < K; kc += KC) {
        __syncthreads();  // previous chunk fully consumed before restaging
        // Cooperative stage of BT[:, kc:kc+KC] into padded LDS panel (16B chunks).
        constexpr int CHUNKS = ncols * (KC / 8);
        for (int idx = threadIdx.x; idx < CHUNKS; idx += blockDim.x) {
            const int c  = idx / (KC / 8);
            const int ck = idx - c * (KC / 8);
            *(float4*)(smem + c * ROWB + ck * 16) =
                *(const float4*)(BT + (size_t)c * K + kc + ck * 8);
        }
        __syncthreads();

        if (active) {
            const float* arow = A + (size_t)(m0 + sub) * K + kc;

            v16bf bcur[NT];
#pragma unroll
            for (int nt = 0; nt < NT; ++nt) {
                bcur[nt] = *(const v16bf*)(bbase + nt * 16 * ROWB);
            }

#pragma unroll
            for (int k = 0; k < KC; k += 32) {
                // A fragment (16x32 bf16): lane holds K = k+16*half+{0..15}
                const float4* ap = (const float4*)(arow + k + 16 * half);
                const float4 a0 = ap[0], a1 = ap[1], a2 = ap[2], a3 = ap[3];
                v16bf af;
                af[0]  = (__bf16)a0.x; af[1]  = (__bf16)a0.y; af[2]  = (__bf16)a0.z; af[3]  = (__bf16)a0.w;
                af[4]  = (__bf16)a1.x; af[5]  = (__bf16)a1.y; af[6]  = (__bf16)a1.z; af[7]  = (__bf16)a1.w;
                af[8]  = (__bf16)a2.x; af[9]  = (__bf16)a2.y; af[10] = (__bf16)a2.z; af[11] = (__bf16)a2.w;
                af[12] = (__bf16)a3.x; af[13] = (__bf16)a3.y; af[14] = (__bf16)a3.z; af[15] = (__bf16)a3.w;

                // Issue next k-step's B-fragment loads before this step's WMMAs.
                v16bf bnxt[NT];
                if (k + 32 < KC) {
#pragma unroll
                    for (int nt = 0; nt < NT; ++nt) {
                        bnxt[nt] = *(const v16bf*)(bbase + nt * 16 * ROWB + (k + 32) * 2);
                    }
                }
#pragma unroll
                for (int nt = 0; nt < NT; ++nt) {
                    acc[nt] = __builtin_amdgcn_wmma_f32_16x16x32_bf16(
                        false, af, false, bcur[nt], (short)0, acc[nt], false, false);
                }
                if (k + 32 < KC) {
#pragma unroll
                    for (int nt = 0; nt < NT; ++nt) bcur[nt] = bnxt[nt];
                }
            }
        }
    }

    if (active) {
        // C/D layout: VGPR j -> row m0 + j + 8*half, col = nt*16 + sub
#pragma unroll
        for (int nt = 0; nt < NT; ++nt) {
            const int col = nt * 16 + sub;
            const float bv = bias ? bias[col] : 0.0f;
#pragma unroll
            for (int j = 0; j < 8; ++j) {
                const int row = m0 + j + 8 * half;
                float v = acc[nt][j] + bv;
                if (ACT) v = gelu_f(v);
                C[(size_t)row * ncols + col] = v;
            }
        }
    }
}

// ----------------------------- GCN aggregation -----------------------------

// a[i,f] = t[i,f] * dinv[i]^2 + bias[f]   (self-loop term + conv bias)
__global__ void k_agg_init(const float* __restrict__ t, const float* __restrict__ dinv,
                           const float* __restrict__ bias, float* __restrict__ a, int total) {
    int i = blockIdx.x * blockDim.x + threadIdx.x;
    for (; i < total; i += gridDim.x * blockDim.x) {
        const int f = i & 127;
        const int node = i >> 7;
        const float w = dinv[node];
        a[i] = t[i] * w * w + bias[f];
    }
}

// One wave per edge: gather 128-float src row (float4/lane), scatter-add into dst row.
// Activations (51 MB) live in the 192 MB L2, so atomics resolve near the L2 atomic units.
__global__ void k_agg_edges(const float* __restrict__ t, const float* __restrict__ dinv,
                            const int* __restrict__ src, const int* __restrict__ dst,
                            float* __restrict__ a, int E) {
    const int lane = threadIdx.x & 31;
    int wid = (blockIdx.x * blockDim.x + threadIdx.x) >> 5;
    const int nw = (gridDim.x * blockDim.x) >> 5;
    for (int e = wid; e < E; e += nw) {
        const int s = src[e];
        const int d = dst[e];
        if (e + nw < E) {  // prefetch next gather row (global_prefetch_b8)
            __builtin_prefetch(t + (size_t)src[e + nw] * 128 + lane * 4, 0, 1);
        }
        const float w = dinv[s] * dinv[d];
        const float4 v = *(const float4*)(t + (size_t)s * 128 + lane * 4);
        float* outp = a + (size_t)d * 128 + lane * 4;
        atomic_add_agent(outp + 0, v.x * w);
        atomic_add_agent(outp + 1, v.y * w);
        atomic_add_agent(outp + 2, v.z * w);
        atomic_add_agent(outp + 3, v.w * w);
    }
}

// ----------------------------- batch norm -----------------------------

// stats[0..127] = sum, stats[128..255] = sum of squares (caller zeroes first)
__global__ void k_bn_stats(const float* __restrict__ a, float* __restrict__ stats, int nrows) {
    __shared__ float ssum[256];
    __shared__ float ssq[256];
    const int tid = threadIdx.x;
    const int f = tid & 127;
    const int p = tid >> 7;
    float s = 0.f, q = 0.f;
    for (int r = blockIdx.x * 2 + p; r < nrows; r += gridDim.x * 2) {
        const float v = a[(size_t)r * 128 + f];
        s += v;
        q += v * v;
    }
    ssum[tid] = s;
    ssq[tid]  = q;
    __syncthreads();
    if (tid < 128) {
        atomic_add_agent(&stats[f],       ssum[tid] + ssum[tid + 128]);
        atomic_add_agent(&stats[128 + f], ssq[tid]  + ssq[tid + 128]);
    }
}

// out = gelu( (a - mean) * rsqrt(var+eps) * gamma + beta ) + ident
__global__ void k_bn_apply_gelu_res(const float* __restrict__ a, const float* __restrict__ stats,
                                    const float* __restrict__ gamma, const float* __restrict__ beta,
                                    const float* __restrict__ ident, float* __restrict__ out,
                                    int total, float invN) {
    int i = blockIdx.x * blockDim.x + threadIdx.x;
    for (; i < total; i += gridDim.x * blockDim.x) {
        const int f = i & 127;
        const float mean = stats[f] * invN;
        const float var  = stats[128 + f] * invN - mean * mean;
        const float istd = rsqrtf(var + BN_EPS_F);
        const float v = (a[i] - mean) * istd * gamma[f] + beta[f];
        out[i] = gelu_f(v) + ident[i];
    }
}

// ----------------------------- driver -----------------------------

extern "C" void kernel_launch(void* const* d_in, const int* in_sizes, int n_in,
                              void* d_out, int out_size, void* d_ws, size_t ws_size,
                              hipStream_t stream) {
    (void)n_in; (void)out_size; (void)ws_size;

    const float* x        = (const float*)d_in[0];
    const float* W_reduce = (const float*)d_in[1];
    const float* b_reduce = (const float*)d_in[2];
    const float* W1       = (const float*)d_in[3];
    const float* b1       = (const float*)d_in[4];
    const float* gamma1   = (const float*)d_in[5];
    const float* beta1    = (const float*)d_in[6];
    const float* W2       = (const float*)d_in[7];
    const float* b2       = (const float*)d_in[8];
    const float* gamma2   = (const float*)d_in[9];
    const float* beta2    = (const float*)d_in[10];
    const float* W_lin    = (const float*)d_in[11];
    const float* b_lin    = (const float*)d_in[12];
    const int*   edges    = (const int*)d_in[13];

    const int N = in_sizes[0] / 256;   // 100000
    const int E = in_sizes[13] / 2;    // 1600000
    const int* srcp = edges;
    const int* dstp = edges + E;

    // Workspace layout (all offsets keep >=32B alignment for wide loads)
    float* ws = (float*)d_ws;
    const size_t NF = (size_t)N * 128;
    float* buf0 = ws;                  // h0 -> t2 -> h2
    float* buf1 = ws + NF;             // t1 -> h1
    float* buf2 = ws + 2 * NF;         // aggregation accumulator
    float* dinv = ws + 3 * NF;         // [N]
    float* stats = dinv + (size_t)((N + 255) & ~255);  // [256]
    __bf16* WTr = (__bf16*)(stats + 256);      // [128][256]
    __bf16* WT1 = WTr + 128 * 256;             // [128][128]
    __bf16* WT2 = WT1 + 128 * 128;             // [128][128]
    __bf16* WTl = WT2 + 128 * 128;             // [64][128]

    const int tiles = (N + 15) / 16;
    const int gemm_blocks = (tiles + 3) / 4;   // 4 waves (128 threads) per block
    const int total = N * 128;
    const float invN = 1.0f / (float)N;

    const size_t smem8 = (size_t)128 * (2 * 128 + 16);  // 34816 B, NT=8 panels
    const size_t smem4 = (size_t)64  * (2 * 128 + 16);  // 17408 B, NT=4 panel

    // --- prep: bf16 transposed weights, degrees ---
    k_cvt_transpose_bf16<<<128, 256, 0, stream>>>(W_reduce, WTr, 256, 128);
    k_cvt_transpose_bf16<<<64,  256, 0, stream>>>(W1, WT1, 128, 128);
    k_cvt_transpose_bf16<<<64,  256, 0, stream>>>(W2, WT2, 128, 128);
    k_cvt_transpose_bf16<<<32,  256, 0, stream>>>(W_lin, WTl, 128, 64);
    k_fill<<<256, 256, 0, stream>>>(dinv, 1.0f, N);               // self-loop degree
    k_deg_count<<<2048, 256, 0, stream>>>(dstp, dinv, E);
    k_rsqrt_inplace<<<256, 256, 0, stream>>>(dinv, N);

    // --- h0 = gelu(x @ W_reduce + b_reduce) ---
    k_gemm_wmma<8, 1><<<gemm_blocks, 128, smem8, stream>>>(x, WTr, b_reduce, buf0, N, 256);

    // --- layer 1: gcn_conv -> bn -> gelu -> +identity ---
    k_gemm_wmma<8, 0><<<gemm_blocks, 128, smem8, stream>>>(buf0, WT1, nullptr, buf1, N, 128);
    k_agg_init<<<2048, 256, 0, stream>>>(buf1, dinv, b1, buf2, total);
    k_agg_edges<<<8192, 256, 0, stream>>>(buf1, dinv, srcp, dstp, buf2, E);
    k_fill<<<1, 256, 0, stream>>>(stats, 0.0f, 256);
    k_bn_stats<<<512, 256, 0, stream>>>(buf2, stats, N);
    k_bn_apply_gelu_res<<<2048, 256, 0, stream>>>(buf2, stats, gamma1, beta1,
                                                  buf0, buf1, total, invN);  // buf1 = h1

    // --- layer 2 ---
    k_gemm_wmma<8, 0><<<gemm_blocks, 128, smem8, stream>>>(buf1, WT2, nullptr, buf0, N, 128);
    k_agg_init<<<2048, 256, 0, stream>>>(buf0, dinv, b2, buf2, total);
    k_agg_edges<<<8192, 256, 0, stream>>>(buf0, dinv, srcp, dstp, buf2, E);
    k_fill<<<1, 256, 0, stream>>>(stats, 0.0f, 256);
    k_bn_stats<<<512, 256, 0, stream>>>(buf2, stats, N);
    k_bn_apply_gelu_res<<<2048, 256, 0, stream>>>(buf2, stats, gamma2, beta2,
                                                  buf1, buf0, total, invN);  // buf0 = h2

    // --- out = h2 @ W_lin + b_lin ---
    k_gemm_wmma<4, 0><<<gemm_blocks, 128, smem4, stream>>>(buf0, WTl, b_lin, (float*)d_out, N, 128);
}
